// RadianceField_32238024524538
// MI455X (gfx1250) — compile-verified, matrix operand order checked by reading, not run on hardware
//
#include <hip/hip_runtime.h>
#include <cstddef>

// ---------------------------------------------------------------------------
// NeRF fused forward for MI455X (gfx1250): all GEMMs via v_wmma_f32_16x16x32_f16
// v2: each wave owns 2 M-tiles (32 rows), waves split N-tiles -> every B
// fragment feeds two independent WMMA chains (half the L2 traffic, no
// accumulator RAW hazard between consecutive WMMAs).
// ---------------------------------------------------------------------------

#define N_PTS 262144
#define ROWS_PER_WG 32
#define LSTR 328            // LDS row stride in halves (16B aligned, bank-friendly)

typedef _Float16 half16 __attribute__((ext_vector_type(16)));
typedef _Float16 half8  __attribute__((ext_vector_type(8)));
typedef float    float8 __attribute__((ext_vector_type(8)));

// Layer table (15 layers):
// 0: layer0  1-4: skip0  5: layer5  6-8: skip1  9: sigma  10: layer9  11-13: skip2  14: rgb
__constant__ __device__ const int c_Kin[15]  = {60,256,256,256,256, 316,256,256,256, 256, 280,128,128,128, 128};
__constant__ __device__ const int c_Nout[15] = {256,256,256,256,256, 256,256,256,256,   1, 128,128,128,128,   3};
__constant__ __device__ const int c_KB[15]   = {2,8,8,8,8, 10,8,8,8, 8, 9,4,4,4, 4};
// cumulative fragment bases (frag = 1KB = 512 halves)
__constant__ __device__ const int c_FB[15]   = {0,32,160,288,416, 544,704,832,960, 1088, 1096,1168,1200,1232, 1264};
#define TOTAL_FRAGS 1268

struct Wts    { const float* w[15]; };
struct Biases { const float* b[15]; };

// ---------------------------------------------------------------------------
// Prep: convert f32 weights to f16 B-fragments (WMMA 16x16x32 layout).
// Per fragment (one 32x16 K-slab x N-tile): lane l holds column n = nt*16+(l&15),
// K values k = kb*32 + (l>>4)*8 + {0..7, 16..23}. 32 bytes per lane.
// ---------------------------------------------------------------------------
__global__ void prep_weights(Wts wts, _Float16* __restrict__ wsf) {
    int tid  = blockIdx.x * blockDim.x + threadIdx.x;
    int frag = tid >> 5;
    int lane = tid & 31;
    if (frag >= TOTAL_FRAGS) return;

    int layer = 0;
    #pragma unroll
    for (int i = 1; i < 15; ++i) layer += (frag >= c_FB[i]) ? 1 : 0;

    const int lf  = frag - c_FB[layer];
    const int KB  = c_KB[layer];
    const int Kin = c_Kin[layer];
    const int Nout= c_Nout[layer];
    const int nt  = lf / KB;
    const int kb  = lf % KB;
    const int n   = nt * 16 + (lane & 15);
    const int khi = (lane >> 4) * 8;
    const float* W = wts.w[layer];

    half8 lo = {}, hv = {};
    #pragma unroll
    for (int j = 0; j < 8; ++j) {
        int k = kb * 32 + khi + j;
        lo[j] = (k < Kin && n < Nout) ? (_Float16)W[k * Nout + n] : (_Float16)0.0f;
    }
    #pragma unroll
    for (int j = 0; j < 8; ++j) {
        int k = kb * 32 + khi + 16 + j;
        hv[j] = (k < Kin && n < Nout) ? (_Float16)W[k * Nout + n] : (_Float16)0.0f;
    }
    _Float16* dst = wsf + (size_t)frag * 512 + lane * 16;
    *(half8*)(dst)     = lo;
    *(half8*)(dst + 8) = hv;
}

__device__ __forceinline__ half16 load_afrag(const _Float16* p) {
    half8 alo = *(const half8*)(p);
    half8 ahi = *(const half8*)(p + 16);
    return __builtin_shufflevector(alo, ahi, 0,1,2,3,4,5,6,7,8,9,10,11,12,13,14,15);
}

// ---------------------------------------------------------------------------
// One fused layer. NT>1: each wave processes BOTH 16-row M-tiles and half the
// N-tiles (B fragment reuse x2, dual independent acc chains). NT==1: each wave
// processes its own 16-row M-tile (sigma/rgb epilogues).
// OUT: 0 = LDS store relu, 1 = LDS store relu+residual,
//      2 = sigma (relu, col0 -> global), 3 = rgb (sigmoid, cols0..2 -> global)
// ---------------------------------------------------------------------------
template <int KB, int NT, int OUT>
__device__ __forceinline__ void layer_gemm(const _Float16* __restrict__ in,
                                           _Float16* __restrict__ outl,
                                           const _Float16* __restrict__ wf,
                                           const float* __restrict__ bias,
                                           float* __restrict__ gout, int grow0) {
    const int lane = threadIdx.x & 31;
    const int wave = threadIdx.x >> 5;
    const int m    = lane & 15;
    const int hi   = lane >> 4;

    if constexpr (NT == 1) {
        // --- single N-tile: per-wave M-tile, global epilogue ---
        half16 A[KB];
        const _Float16* ar = in + (wave * 16 + m) * LSTR + hi * 8;
        #pragma unroll
        for (int kb = 0; kb < KB; ++kb) A[kb] = load_afrag(ar + kb * 32);

        float8 acc = {};
        const half8* bp = (const half8*)(wf + lane * 16);
        #pragma unroll
        for (int kb = 0; kb < KB; ++kb) {
            half8 blo = bp[0], bhi = bp[1];
            bp += 64;
            half16 B = __builtin_shufflevector(blo, bhi, 0,1,2,3,4,5,6,7,8,9,10,11,12,13,14,15);
            acc = __builtin_amdgcn_wmma_f32_16x16x32_f16(
                false, A[kb], false, B, (short)0, acc, false, false);
        }
        if constexpr (OUT == 2) { // sigma: relu, only column 0
            if (m == 0) {
                const float bv = bias[0];
                #pragma unroll
                for (int r = 0; r < 8; ++r) {
                    float v = acc[r] + bv;
                    v = v > 0.0f ? v : 0.0f;
                    gout[grow0 + wave * 16 + hi * 8 + r] = v;
                }
            }
        } else { // rgb: sigmoid, columns 0..2
            if (m < 3) {
                const float bv = bias[m];
                #pragma unroll
                for (int r = 0; r < 8; ++r) {
                    float v = acc[r] + bv;
                    v = 1.0f / (1.0f + __expf(-v));
                    gout[(size_t)(grow0 + wave * 16 + hi * 8 + r) * 3 + m] = v;
                }
            }
        }
    } else {
        // --- split-N: wave owns N-tiles [wave*NT/2, ...), both M-tiles ---
        constexpr int NTW = NT / 2;
        const int ntBase = wave * NTW;

        half16 A0[KB], A1[KB];
        const _Float16* ar0 = in + m * LSTR + hi * 8;          // rows 0..15
        const _Float16* ar1 = in + (16 + m) * LSTR + hi * 8;   // rows 16..31
        #pragma unroll
        for (int kb = 0; kb < KB; ++kb) {
            A0[kb] = load_afrag(ar0 + kb * 32);
            A1[kb] = load_afrag(ar1 + kb * 32);
        }

        #pragma unroll 1
        for (int t = 0; t < NTW; ++t) {
            const int nt = ntBase + t;
            float8 acc0 = {}, acc1 = {};
            const half8* bp = (const half8*)(wf + (size_t)(nt * KB) * 512 + lane * 16);
            #pragma unroll
            for (int kb = 0; kb < KB; ++kb) {
                half8 blo = bp[0], bhi = bp[1];
                bp += 64; // next fragment (512 halves)
                half16 B = __builtin_shufflevector(blo, bhi, 0,1,2,3,4,5,6,7,8,9,10,11,12,13,14,15);
                // two independent accumulator chains sharing one B fragment
                acc0 = __builtin_amdgcn_wmma_f32_16x16x32_f16(
                    false, A0[kb], false, B, (short)0, acc0, false, false);
                acc1 = __builtin_amdgcn_wmma_f32_16x16x32_f16(
                    false, A1[kb], false, B, (short)0, acc1, false, false);
            }
            // bp now points at the next N-tile's first fragment: prefetch it
            if (t + 1 < NTW) __builtin_prefetch((const void*)bp, 0, 1);

            const int col = nt * 16 + m;
            const float bv = bias[col];
            #pragma unroll
            for (int r = 0; r < 8; ++r) {
                const int rl   = hi * 8 + r;
                const int idx0 = rl * LSTR + col;
                const int idx1 = (16 + rl) * LSTR + col;
                float v0 = acc0[r] + bv;
                float v1 = acc1[r] + bv;
                v0 = v0 > 0.0f ? v0 : 0.0f;
                v1 = v1 > 0.0f ? v1 : 0.0f;
                if (OUT == 1) {
                    v0 += (float)in[idx0];
                    v1 += (float)in[idx1];
                }
                outl[idx0] = (_Float16)v0;
                outl[idx1] = (_Float16)v1;
            }
        }
    }
}

// ---------------------------------------------------------------------------
// Fused NeRF forward: encode -> layer0 -> skip0 x4 -> [h,e_x] -> layer5 ->
// skip1 x3 -> sigma -> [h,e_d] -> layer9 -> skip2 x3 -> rgb
// ---------------------------------------------------------------------------
__global__ __launch_bounds__(64) void nerf_fused(const float* __restrict__ x,
                                                 const float* __restrict__ d,
                                                 const _Float16* __restrict__ wsf,
                                                 float* __restrict__ out,
                                                 Biases bs) {
    __shared__ _Float16 bufA[ROWS_PER_WG * LSTR];
    __shared__ _Float16 bufB[ROWS_PER_WG * LSTR];
    __shared__ _Float16 exS[ROWS_PER_WG * 64];
    __shared__ _Float16 edS[ROWS_PER_WG * 32];

    const int tid  = threadIdx.x;
    const int row0 = blockIdx.x * ROWS_PER_WG;

    // ---- positional encoding -> bufA (layer0 input) + stashes ----
    if (tid < 32) {
        const int r = tid, g = row0 + r;
        float px[3] = {x[g * 3 + 0], x[g * 3 + 1], x[g * 3 + 2]};
        _Float16* br = bufA + r * LSTR;
        _Float16* er = exS + r * 64;
        #pragma unroll
        for (int c = 0; c < 3; ++c) {
            float f = 3.14159265358979f;
            #pragma unroll
            for (int i = 0; i < 10; ++i) {
                float a = px[c] * f;
                float s = __sinf(a), co = __cosf(a);
                int o = c * 20 + i * 2;
                br[o] = (_Float16)s;  br[o + 1] = (_Float16)co;
                er[o] = (_Float16)s;  er[o + 1] = (_Float16)co;
                f *= 2.0f;
            }
        }
        #pragma unroll
        for (int o = 60; o < 64; ++o) { br[o] = (_Float16)0.0f; er[o] = (_Float16)0.0f; }
    } else {
        const int r = tid - 32, g = row0 + r;
        float pd[3] = {d[g * 3 + 0], d[g * 3 + 1], d[g * 3 + 2]};
        _Float16* er = edS + r * 32;
        #pragma unroll
        for (int c = 0; c < 3; ++c) {
            float f = 3.14159265358979f;
            #pragma unroll
            for (int i = 0; i < 4; ++i) {
                float a = pd[c] * f;
                int o = c * 8 + i * 2;
                er[o] = (_Float16)__sinf(a);
                er[o + 1] = (_Float16)__cosf(a);
                f *= 2.0f;
            }
        }
        #pragma unroll
        for (int o = 24; o < 32; ++o) er[o] = (_Float16)0.0f;
    }
    __syncthreads();

    // layer0: e_x(64) -> 256
    layer_gemm<2, 16, 0>(bufA, bufB, wsf + (size_t)c_FB[0] * 512, bs.b[0], nullptr, 0);
    __syncthreads();
    // skip0 x4
    layer_gemm<8, 16, 1>(bufB, bufA, wsf + (size_t)c_FB[1] * 512, bs.b[1], nullptr, 0);
    __syncthreads();
    layer_gemm<8, 16, 1>(bufA, bufB, wsf + (size_t)c_FB[2] * 512, bs.b[2], nullptr, 0);
    __syncthreads();
    layer_gemm<8, 16, 1>(bufB, bufA, wsf + (size_t)c_FB[3] * 512, bs.b[3], nullptr, 0);
    __syncthreads();
    layer_gemm<8, 16, 1>(bufA, bufB, wsf + (size_t)c_FB[4] * 512, bs.b[4], nullptr, 0);
    __syncthreads();
    // concat e_x into bufB cols 256..319 (60 real + 4 zero pad -> K=320)
    for (int i = tid; i < ROWS_PER_WG * 64; i += 64) {
        int r = i >> 6, c = i & 63;
        bufB[r * LSTR + 256 + c] = exS[i];
    }
    __syncthreads();
    // layer5: 320 -> 256
    layer_gemm<10, 16, 0>(bufB, bufA, wsf + (size_t)c_FB[5] * 512, bs.b[5], nullptr, 0);
    __syncthreads();
    // skip1 x3
    layer_gemm<8, 16, 1>(bufA, bufB, wsf + (size_t)c_FB[6] * 512, bs.b[6], nullptr, 0);
    __syncthreads();
    layer_gemm<8, 16, 1>(bufB, bufA, wsf + (size_t)c_FB[7] * 512, bs.b[7], nullptr, 0);
    __syncthreads();
    layer_gemm<8, 16, 1>(bufA, bufB, wsf + (size_t)c_FB[8] * 512, bs.b[8], nullptr, 0);
    __syncthreads();
    // sigma: 256 -> 1 (relu), straight to global (out[3N .. 4N))
    layer_gemm<8, 1, 2>(bufB, nullptr, wsf + (size_t)c_FB[9] * 512, bs.b[9],
                        out + (size_t)3 * N_PTS, row0);
    // concat e_d into bufB cols 256..287 (24 real + 8 zero pad -> K=288)
    for (int i = tid; i < ROWS_PER_WG * 32; i += 64) {
        int r = i >> 5, c = i & 31;
        bufB[r * LSTR + 256 + c] = edS[i];
    }
    __syncthreads();
    // layer9: 288 -> 128
    layer_gemm<9, 8, 0>(bufB, bufA, wsf + (size_t)c_FB[10] * 512, bs.b[10], nullptr, 0);
    __syncthreads();
    // skip2 x3
    layer_gemm<4, 8, 1>(bufA, bufB, wsf + (size_t)c_FB[11] * 512, bs.b[11], nullptr, 0);
    __syncthreads();
    layer_gemm<4, 8, 1>(bufB, bufA, wsf + (size_t)c_FB[12] * 512, bs.b[12], nullptr, 0);
    __syncthreads();
    layer_gemm<4, 8, 1>(bufA, bufB, wsf + (size_t)c_FB[13] * 512, bs.b[13], nullptr, 0);
    __syncthreads();
    // rgb: 128 -> 3 (sigmoid), straight to global (out[0 .. 3N))
    layer_gemm<4, 1, 3>(bufB, nullptr, wsf + (size_t)c_FB[14] * 512, bs.b[14], out, row0);
}

// ---------------------------------------------------------------------------
extern "C" void kernel_launch(void* const* d_in, const int* in_sizes, int n_in,
                              void* d_out, int out_size, void* d_ws, size_t ws_size,
                              hipStream_t stream) {
    const float* x = (const float*)d_in[0];
    const float* d = (const float*)d_in[1];

    // Detect params flattening order: jax-sorted keys vs. insertion order.
    // sorted:   layer0@2 layer5@4 layer9@6 rgb@8 sigma@10 skip0@12 skip1@20 skip2@26
    // insertion:layer0@2 skip0@4 layer5@12 skip1@14 sigma@20 layer9@22 skip2@24 rgb@30
    bool sorted_order = (n_in > 4) && (in_sizes[4] == 316 * 256);
    int mapW[15];
    if (sorted_order) {
        const int m[15] = {2, 12,14,16,18, 4, 20,22,24, 10, 6, 26,28,30, 8};
        for (int i = 0; i < 15; ++i) mapW[i] = m[i];
    } else {
        const int m[15] = {2, 4,6,8,10, 12, 14,16,18, 20, 22, 24,26,28, 30};
        for (int i = 0; i < 15; ++i) mapW[i] = m[i];
    }

    Wts wts; Biases bs;
    for (int i = 0; i < 15; ++i) {
        wts.w[i] = (const float*)d_in[mapW[i]];
        bs.b[i]  = (const float*)d_in[mapW[i] + 1];
    }

    _Float16* wsf = (_Float16*)d_ws;
    (void)ws_size; (void)out_size;

    // 1) repack weights into f16 WMMA B-fragments in workspace (deterministic)
    {
        int threads = TOTAL_FRAGS * 32;
        int blk = 256;
        prep_weights<<<(threads + blk - 1) / blk, blk, 0, stream>>>(wts, wsf);
    }
    // 2) fused forward
    nerf_fused<<<N_PTS / ROWS_PER_WG, 64, 0, stream>>>(x, d, wsf, (float*)d_out, bs);
}
